// WesnothTransformer_51608327028842
// MI455X (gfx1250) — matrix operand order repaired and checked by reading, not compiled, not run on hardware
//
#include <hip/hip_runtime.h>
#include <hip/hip_bf16.h>

// ---------------------------------------------------------------------------
// WesnothTransformer forward for MI455X (gfx1250, wave32, WMMA bf16 path)
// ---------------------------------------------------------------------------

typedef __bf16 bf16_t;
typedef __attribute__((ext_vector_type(16))) __bf16 v16bf;
typedef __attribute__((ext_vector_type(8)))  __bf16 v8bf;
typedef __attribute__((ext_vector_type(8)))  float  v8f;

#define Bn    2
#define Sn    2304          // 48*48 tokens
#define Dn    512
#define NHn   8
#define DKn   64
#define Ln    4
#define DFn   2048
#define Mn    128           // memory slots
#define FHEX  150
#define FHEXP 160
#define FGLB  7

// ---------------- WMMA helpers (layouts per CDNA5 ISA 7.12.2) ----------------

__device__ inline v8f wmma_bf16(v16bf a, v16bf b, v8f c) {
  return __builtin_amdgcn_wmma_f32_16x16x32_bf16(false, a, false, b, (short)0, c, false, false);
}

// A fragment 16x32 bf16, A row-major, ptr at (row0, k0). lane<16: row=lane,
// elems 0..7 = K0..7, 8..15 = K16..23 ; lane>=16: row=lane-16, K{8..15,24..31}.
__device__ inline v16bf frag_a(const bf16_t* p0, int lda, int lane) {
  const bf16_t* p = p0 + (long)(lane & 15) * lda + ((lane >> 4) << 3);
  v8bf lo = *(const v8bf*)p;
  v8bf hi = *(const v8bf*)(p + 16);
  v16bf f;
#pragma unroll
  for (int i = 0; i < 8; ++i) { f[i] = lo[i]; f[i + 8] = hi[i]; }
  return f;
}

// B fragment 32x16 bf16 from Bt stored [N,K] row-major, ptr at (col0, k0).
// lane<16: col=lane, elems 0..7 = K0..7, 8..15 = K16..23 ; lane>=16: col=lane-16, K{8..15,24..31}.
__device__ inline v16bf frag_b(const bf16_t* p0, int ldb, int lane) {
  const bf16_t* p = p0 + (long)(lane & 15) * ldb + ((lane >> 4) << 3);
  v8bf lo = *(const v8bf*)p;
  v8bf hi = *(const v8bf*)(p + 16);
  v16bf f;
#pragma unroll
  for (int i = 0; i < 8; ++i) { f[i] = lo[i]; f[i + 8] = hi[i]; }
  return f;
}

__device__ inline float rmax16(float v) {
#pragma unroll
  for (int m = 1; m < 16; m <<= 1) v = fmaxf(v, __shfl_xor(v, m, 32));
  return v;
}
__device__ inline float rsum16(float v) {
#pragma unroll
  for (int m = 1; m < 16; m <<= 1) v += __shfl_xor(v, m, 32);
  return v;
}

// ------------------------------ GEMM kernel ---------------------------------
// C[M,N] = act( A[M,K]_bf16 @ Bt[N,K]_bf16^T + bias + rowvec + res )
// block = 256 thr = 8 waves ; block tile 128x64 ; wave tile 32x32 (2x2 WMMA)
#define GF_BIAS   1
#define GF_RES    2
#define GF_GELU   4
#define GF_SIGM   8
#define GF_ROWVEC 16

__global__ __launch_bounds__(256) void k_gemm(
    const bf16_t* __restrict__ A, int lda,
    const bf16_t* __restrict__ Bt, int ldb,
    const float* __restrict__ bias,
    const float* __restrict__ res, int ldr,
    const float* __restrict__ rowvec, int rpb,
    float* __restrict__ Cf, bf16_t* __restrict__ Cb, int ldc,
    int Mdim, int Ndim, int Kdim, int flags) {
  int lane = threadIdx.x & 31;
  int wv = threadIdx.x >> 5;
  int wm = wv & 3, wn = wv >> 2;
  int row0 = blockIdx.y * 128 + wm * 32;
  int col0 = blockIdx.x * 64 + wn * 32;
  if (row0 >= Mdim || col0 >= Ndim) return;

  v8f acc[2][2];
#pragma unroll
  for (int i = 0; i < 2; ++i)
#pragma unroll
    for (int j = 0; j < 2; ++j)
#pragma unroll
      for (int r = 0; r < 8; ++r) acc[i][j][r] = 0.f;

  const bf16_t* Ab = A + (long)row0 * lda;
  const bf16_t* Bb = Bt + (long)col0 * ldb;
  for (int k = 0; k < Kdim; k += 32) {
    v16bf a0 = frag_a(Ab + k, lda, lane);
    v16bf a1 = frag_a(Ab + (long)16 * lda + k, lda, lane);
    v16bf b0 = frag_b(Bb + k, ldb, lane);
    v16bf b1 = frag_b(Bb + (long)16 * ldb + k, ldb, lane);
    acc[0][0] = wmma_bf16(a0, b0, acc[0][0]);
    acc[0][1] = wmma_bf16(a0, b1, acc[0][1]);
    acc[1][0] = wmma_bf16(a1, b0, acc[1][0]);
    acc[1][1] = wmma_bf16(a1, b1, acc[1][1]);
  }

  int nloc = lane & 15;
  int mloc = (lane >> 4) << 3;
#pragma unroll
  for (int i = 0; i < 2; ++i)
#pragma unroll
    for (int j = 0; j < 2; ++j) {
      int n = col0 + 16 * j + nloc;
      float bv = (flags & GF_BIAS) ? bias[n] : 0.f;
#pragma unroll
      for (int r = 0; r < 8; ++r) {
        int m = row0 + 16 * i + mloc + r;
        float v = acc[i][j][r] + bv;
        if (flags & GF_ROWVEC) v += rowvec[(m / rpb) * Ndim + n];
        if (flags & GF_RES)    v += res[(long)m * ldr + n];
        if (flags & GF_GELU)   v = 0.5f * v * (1.f + erff(v * 0.70710678118f));
        if (flags & GF_SIGM)   v = 1.f / (1.f + __expf(-v));
        if (Cf) Cf[(long)m * ldc + n] = v;
        if (Cb) Cb[(long)m * ldc + n] = (bf16_t)v;
      }
    }
}

// --------------------------- flash attention --------------------------------
// Q bf16 [B*Sq, 512] (head offset h*64), K bf16 [B*Skv, 512],
// Vt bf16 [B, 512, Skv]  (V transposed so B-fragments are contiguous),
// mask int32 [B, Skv] or null, O bf16 [B*Sq, 512].
// block = 128 thr = 4 waves ; one wave per 16-row query tile ; key tiles of 32.
__global__ __launch_bounds__(128) void k_flash(
    const bf16_t* __restrict__ Q, const bf16_t* __restrict__ K,
    const bf16_t* __restrict__ Vt, const int* __restrict__ mask,
    bf16_t* __restrict__ O, int Sq, int Skv, float scale) {
  __shared__ __align__(16) bf16_t pbuf[4][16 * 32];
  int lane = threadIdx.x & 31;
  int wv = threadIdx.x >> 5;
  int qt = blockIdx.x * 4 + wv;
  int h = blockIdx.y, b = blockIdx.z;
  int qrow0 = qt * 16;
  if (qrow0 >= Sq) return;

  const bf16_t* Qb = Q + ((long)b * Sq + qrow0) * Dn + h * DKn;
  const bf16_t* Kb = K + (long)b * Skv * Dn + h * DKn;
  const bf16_t* Vb = Vt + ((long)b * Dn + h * DKn) * Skv;

  v16bf qa0 = frag_a(Qb, Dn, lane);        // d 0..31
  v16bf qa1 = frag_a(Qb + 32, Dn, lane);   // d 32..63

  v8f o[4];
  float mrow[8], lrow[8];
#pragma unroll
  for (int f = 0; f < 4; ++f)
#pragma unroll
    for (int r = 0; r < 8; ++r) o[f][r] = 0.f;
#pragma unroll
  for (int r = 0; r < 8; ++r) { mrow[r] = -1e30f; lrow[r] = 0.f; }

  int nloc = lane & 15;
  int mloc = (lane >> 4) << 3;
  bf16_t* pb = &pbuf[wv][0];

  for (int kt = 0; kt < Skv; kt += 32) {
    v8f s[2];
#pragma unroll
    for (int j = 0; j < 2; ++j) {
      const bf16_t* Kj = Kb + (long)(kt + 16 * j) * Dn;
      v16bf b0 = frag_b(Kj, Dn, lane);
      v16bf b1 = frag_b(Kj + 32, Dn, lane);
      v8f z;
#pragma unroll
      for (int r = 0; r < 8; ++r) z[r] = 0.f;
      z = wmma_bf16(qa0, b0, z);
      s[j] = wmma_bf16(qa1, b1, z);
      bool mz = false;
      if (mask) mz = (mask[(long)b * Skv + kt + 16 * j + nloc] == 0);
#pragma unroll
      for (int r = 0; r < 8; ++r) s[j][r] = mz ? -1e9f : s[j][r] * scale;
    }
    float alpha[8];
#pragma unroll
    for (int r = 0; r < 8; ++r) {
      float mnew = rmax16(fmaxf(s[0][r], s[1][r]));
      float mc = fmaxf(mrow[r], mnew);
      alpha[r] = __expf(mrow[r] - mc);
      mrow[r] = mc;
      float p0 = __expf(s[0][r] - mc);
      float p1 = __expf(s[1][r] - mc);
      s[0][r] = p0; s[1][r] = p1;
      lrow[r] = lrow[r] * alpha[r] + rsum16(p0 + p1);
    }
#pragma unroll
    for (int f = 0; f < 4; ++f)
#pragma unroll
      for (int r = 0; r < 8; ++r) o[f][r] *= alpha[r];
    // stage P (16x32) into LDS, re-layout C-fragment -> A-fragment
#pragma unroll
    for (int r = 0; r < 8; ++r) {
      pb[(mloc + r) * 32 + nloc] = (bf16_t)s[0][r];
      pb[(mloc + r) * 32 + 16 + nloc] = (bf16_t)s[1][r];
    }
    v16bf pa = frag_a(pb, 32, lane);
#pragma unroll
    for (int f = 0; f < 4; ++f) {
      v16bf vbf = frag_b(Vb + (long)(16 * f) * Skv + kt, Skv, lane);
      o[f] = wmma_bf16(pa, vbf, o[f]);
    }
  }

  bf16_t* Ob = O + ((long)b * Sq + qrow0) * Dn + h * DKn;
#pragma unroll
  for (int r = 0; r < 8; ++r) {
    float inv = 1.f / lrow[r];
#pragma unroll
    for (int f = 0; f < 4; ++f)
      Ob[(long)(mloc + r) * Dn + 16 * f + nloc] = (bf16_t)(o[f][r] * inv);
  }
}

// --------------------------- small helper kernels ---------------------------

// weight f32 [K,N] -> bf16 [N,Kp] (transpose + zero-pad K)
__global__ void k_wconv(const float* __restrict__ W, bf16_t* __restrict__ Wt,
                        int Kdim, int Ndim, int Kp) {
  long idx = (long)blockIdx.x * 256 + threadIdx.x;
  long total = (long)Ndim * Kp;
  if (idx >= total) return;
  int n = idx / Kp, k = idx % Kp;
  Wt[idx] = (k < Kdim) ? (bf16_t)W[(long)k * Ndim + n] : (bf16_t)0.f;
}

// activation f32 [M,K] -> bf16 [M,Kp] (zero-pad K)
__global__ void k_aconv(const float* __restrict__ X, bf16_t* __restrict__ Y,
                        int Mrows, int Kin, int Kp) {
  long idx = (long)blockIdx.x * 256 + threadIdx.x;
  long total = (long)Mrows * Kp;
  if (idx >= total) return;
  int r = idx / Kp, k = idx % Kp;
  Y[idx] = (k < Kin) ? (bf16_t)X[(long)r * Kin + k] : (bf16_t)0.f;
}

// bf16 [Bc, R, C] -> [Bc, C, R]
__global__ void k_btrans(const bf16_t* __restrict__ X, bf16_t* __restrict__ Xt,
                         int Bc, int R, int C) {
  long idx = (long)blockIdx.x * 256 + threadIdx.x;
  long total = (long)Bc * R * C;
  if (idx >= total) return;
  int c = idx % C;
  long t = idx / C;
  int r = t % R;
  int b = t / R;
  Xt[((long)b * C + c) * R + r] = X[idx];
}

// glob[b,n] = gf[b,:] @ glob_w + glob_b
__global__ void k_glob(const float* __restrict__ gf, const float* __restrict__ W,
                       const float* __restrict__ bias, float* __restrict__ out) {
  int idx = blockIdx.x * 256 + threadIdx.x;
  if (idx >= Bn * Dn) return;
  int b = idx >> 9, n = idx & 511;
  float acc = bias[n];
#pragma unroll
  for (int k = 0; k < FGLB; ++k) acc += gf[b * FGLB + k] * W[k * Dn + n];
  out[idx] = acc;
}

// LayerNorm over 512 cols; writes f32 and bf16 copies. block=256, 1 row/block.
__global__ __launch_bounds__(256) void k_ln(
    const float* __restrict__ X, const float* __restrict__ g,
    const float* __restrict__ bta, float* __restrict__ Yf,
    bf16_t* __restrict__ Yb) {
  __shared__ float red[256];
  int row = blockIdx.x, t = threadIdx.x;
  const float* x = X + (long)row * Dn;
  float v0 = x[t], v1 = x[t + 256];
  red[t] = v0 + v1;
  __syncthreads();
  for (int w = 128; w > 0; w >>= 1) { if (t < w) red[t] += red[t + w]; __syncthreads(); }
  float mean = red[0] * (1.f / Dn);
  __syncthreads();
  float d0 = v0 - mean, d1 = v1 - mean;
  red[t] = d0 * d0 + d1 * d1;
  __syncthreads();
  for (int w = 128; w > 0; w >>= 1) { if (t < w) red[t] += red[t + w]; __syncthreads(); }
  float rstd = rsqrtf(red[0] * (1.f / Dn) + 1e-5f);
  float y0 = d0 * rstd * g[t] + bta[t];
  float y1 = d1 * rstd * g[t + 256] + bta[t + 256];
  long o = (long)row * Dn;
  if (Yf) { Yf[o + t] = y0; Yf[o + t + 256] = y1; }
  if (Yb) { Yb[o + t] = (bf16_t)y0; Yb[o + t + 256] = (bf16_t)y1; }
}

// new_memory = LN(gate*mupd + (1-gate)*mem) -> d_out region (f32)
__global__ __launch_bounds__(256) void k_newmem(
    const float* __restrict__ gate, const float* __restrict__ mupd,
    const float* __restrict__ mem, const float* __restrict__ g,
    const float* __restrict__ bta, float* __restrict__ out) {
  __shared__ float red[256];
  int row = blockIdx.x, t = threadIdx.x;
  long o = (long)row * Dn;
  float g0 = gate[o + t], g1 = gate[o + t + 256];
  float v0 = g0 * mupd[o + t] + (1.f - g0) * mem[o + t];
  float v1 = g1 * mupd[o + t + 256] + (1.f - g1) * mem[o + t + 256];
  red[t] = v0 + v1;
  __syncthreads();
  for (int w = 128; w > 0; w >>= 1) { if (t < w) red[t] += red[t + w]; __syncthreads(); }
  float mean = red[0] * (1.f / Dn);
  __syncthreads();
  float d0 = v0 - mean, d1 = v1 - mean;
  red[t] = d0 * d0 + d1 * d1;
  __syncthreads();
  for (int w = 128; w > 0; w >>= 1) { if (t < w) red[t] += red[t + w]; __syncthreads(); }
  float rstd = rsqrtf(red[0] * (1.f / Dn) + 1e-5f);
  out[o + t] = d0 * rstd * g[t] + bta[t];
  out[o + t + 256] = d1 * rstd * g[t + 256] + bta[t + 256];
}

// concat [memory | mem_upd] -> bf16 [256,1024]
__global__ void k_cat(const float* __restrict__ mem, const float* __restrict__ mupd,
                      bf16_t* __restrict__ cat) {
  long idx = (long)blockIdx.x * 256 + threadIdx.x;
  if (idx >= (long)Bn * Mn * 1024) return;
  int c = idx & 1023;
  long r = idx >> 10;
  float v = (c < Dn) ? mem[r * Dn + c] : mupd[r * Dn + (c - Dn)];
  cat[idx] = (bf16_t)v;
}

// logits[m] = x[m,:] @ pool_w + pool_b
__global__ void k_logits(const float* __restrict__ X, const float* __restrict__ pw,
                         const float* __restrict__ pbias, float* __restrict__ lg,
                         int rows) {
  int r = blockIdx.x * 256 + threadIdx.x;
  if (r >= rows) return;
  const float* x = X + (long)r * Dn;
  float acc = pbias[0];
  for (int d = 0; d < Dn; ++d) acc += x[d] * pw[d];
  lg[r] = acc;
}

// softmax over S + weighted sum -> pooled [B,512]. 1 block/batch.
__global__ __launch_bounds__(256) void k_pool(
    const float* __restrict__ X, const float* __restrict__ lg,
    float* __restrict__ pooled) {
  __shared__ float sw[Sn];
  __shared__ float red[256];
  int b = blockIdx.x, t = threadIdx.x;
  const float* l = lg + (long)b * Sn;
  float mx = -1e30f;
  for (int s = t; s < Sn; s += 256) mx = fmaxf(mx, l[s]);
  red[t] = mx;
  __syncthreads();
  for (int w = 128; w > 0; w >>= 1) { if (t < w) red[t] = fmaxf(red[t], red[t + w]); __syncthreads(); }
  mx = red[0];
  __syncthreads();
  float sum = 0.f;
  for (int s = t; s < Sn; s += 256) { float e = __expf(l[s] - mx); sw[s] = e; sum += e; }
  red[t] = sum;
  __syncthreads();
  for (int w = 128; w > 0; w >>= 1) { if (t < w) red[t] += red[t + w]; __syncthreads(); }
  float inv = 1.f / red[0];
  __syncthreads();
  for (int d = t; d < Dn; d += 256) {
    float acc = 0.f;
    const float* xb = X + (long)b * Sn * Dn + d;
    for (int s = 0; s < Sn; ++s) acc += sw[s] * xb[(long)s * Dn];
    pooled[b * Dn + d] = acc * inv;
  }
}

// all 8 output heads (444 scalars) directly into d_out[0..443]
__global__ void k_heads(const float* __restrict__ pooled,
                        const float* sxw, const float* sxb, const float* syw, const float* syb,
                        const float* txw, const float* txb, const float* tyw, const float* tyb,
                        const float* akw, const float* akb, const float* rcw, const float* rcb,
                        const float* etw, const float* etb, const float* vlw, const float* vlb,
                        float* __restrict__ out) {
  int i = blockIdx.x * blockDim.x + threadIdx.x;
  if (i >= 444) return;
  const float *W, *bias; int ncols, base;
  if      (i < 96)  { W = sxw; bias = sxb; ncols = 48; base = 0; }
  else if (i < 192) { W = syw; bias = syb; ncols = 48; base = 96; }
  else if (i < 288) { W = txw; bias = txb; ncols = 48; base = 192; }
  else if (i < 384) { W = tyw; bias = tyb; ncols = 48; base = 288; }
  else if (i < 400) { W = akw; bias = akb; ncols = 8;  base = 384; }
  else if (i < 440) { W = rcw; bias = rcb; ncols = 20; base = 400; }
  else if (i < 442) { W = etw; bias = etb; ncols = 1;  base = 440; }
  else              { W = vlw; bias = vlb; ncols = 1;  base = 442; }
  int off = i - base, b = off / ncols, c = off % ncols;
  float acc = bias[c];
  for (int d = 0; d < Dn; ++d) acc += pooled[b * Dn + d] * W[d * ncols + c];
  out[i] = acc;
}

// ------------------------------ workspace map -------------------------------
constexpr size_t AL(size_t x) { return (x + 255) & ~size_t(255); }
constexpr size_t F32SEQ = (size_t)Bn * Sn * Dn * 4;   // 9.44 MB
constexpr size_t BFSEQ  = (size_t)Bn * Sn * Dn * 2;   // 4.72 MB
constexpr size_t BFMEM  = (size_t)Bn * Mn * Dn * 2;   // 256 KB
constexpr size_t O_XF      = 0;
constexpr size_t O_TMPF    = O_XF      + AL(F32SEQ);
constexpr size_t O_SEQF    = O_TMPF    + AL(F32SEQ);
constexpr size_t O_XB      = O_SEQF    + AL(F32SEQ);
constexpr size_t O_SEQB    = O_XB      + AL(BFSEQ);
constexpr size_t O_QB      = O_SEQB    + AL(BFSEQ);
constexpr size_t O_KB      = O_QB      + AL(BFSEQ);
constexpr size_t O_VB      = O_KB      + AL(BFSEQ);
constexpr size_t O_VTB     = O_VB      + AL(BFSEQ);
constexpr size_t O_ATTNB   = O_VTB     + AL(BFSEQ);
constexpr size_t O_HB      = O_ATTNB   + AL(BFSEQ);
constexpr size_t O_MAPB    = O_HB      + AL((size_t)Bn * Sn * DFn * 2);
constexpr size_t O_MEMB    = O_MAPB    + AL((size_t)Bn * Sn * FHEXP * 2);
constexpr size_t O_QMEMB   = O_MEMB    + AL(BFMEM);
constexpr size_t O_KMEMB   = O_QMEMB   + AL(BFMEM);
constexpr size_t O_VMEMB   = O_KMEMB   + AL(BFMEM);
constexpr size_t O_VTMEMB  = O_VMEMB   + AL(BFMEM);
constexpr size_t O_MUPATT  = O_VTMEMB  + AL(BFMEM);
constexpr size_t O_MUPDF   = O_MUPATT  + AL(BFMEM);
constexpr size_t O_GATEF   = O_MUPDF   + AL((size_t)Bn * Mn * Dn * 4);
constexpr size_t O_CATB    = O_GATEF   + AL((size_t)Bn * Mn * Dn * 4);
constexpr size_t O_GLOBF   = O_CATB    + AL((size_t)Bn * Mn * 1024 * 2);
constexpr size_t O_LOGITS  = O_GLOBF   + AL((size_t)Bn * Dn * 4);
constexpr size_t O_POOLED  = O_LOGITS  + AL((size_t)Bn * Sn * 4);
constexpr size_t O_WHEXT   = O_POOLED  + AL((size_t)Bn * Dn * 4);
constexpr size_t O_WMQT    = O_WHEXT   + AL((size_t)Dn * FHEXP * 2);
constexpr size_t O_WMKT    = O_WMQT    + AL((size_t)Dn * Dn * 2);
constexpr size_t O_WMVT    = O_WMKT    + AL((size_t)Dn * Dn * 2);
constexpr size_t O_WMOT    = O_WMVT    + AL((size_t)Dn * Dn * 2);
constexpr size_t O_WGATET  = O_WMOT    + AL((size_t)Dn * Dn * 2);
constexpr size_t O_WLQT    = O_WGATET  + AL((size_t)Dn * 1024 * 2);
constexpr size_t O_WLKT    = O_WLQT    + AL((size_t)Dn * Dn * 2);
constexpr size_t O_WLVT    = O_WLKT    + AL((size_t)Dn * Dn * 2);
constexpr size_t O_WLOT    = O_WLVT    + AL((size_t)Dn * Dn * 2);
constexpr size_t O_WF1T    = O_WLOT    + AL((size_t)Dn * Dn * 2);
constexpr size_t O_WF2T    = O_WF1T    + AL((size_t)DFn * Dn * 2);
constexpr size_t O_END     = O_WF2T    + AL((size_t)Dn * DFn * 2);

// ------------------------------- host driver --------------------------------

static void gemm(hipStream_t st, const bf16_t* A, int lda, const bf16_t* Bt, int ldb,
                 const float* bias, const float* res, int ldr,
                 const float* rowvec, int rpb,
                 float* Cf, bf16_t* Cb, int ldc,
                 int M, int N, int K, int flags) {
  dim3 g(N / 64, M / 128);
  k_gemm<<<g, 256, 0, st>>>(A, lda, Bt, ldb, bias, res, ldr, rowvec, rpb,
                            Cf, Cb, ldc, M, N, K, flags);
}

static void wconv(hipStream_t st, const float* W, bf16_t* Wt, int K, int N, int Kp) {
  long total = (long)N * Kp;
  k_wconv<<<(total + 255) / 256, 256, 0, st>>>(W, Wt, K, N, Kp);
}

extern "C" void kernel_launch(void* const* d_in, const int* in_sizes, int n_in,
                              void* d_out, int out_size, void* d_ws, size_t ws_size,
                              hipStream_t stream) {
  (void)in_sizes; (void)n_in; (void)out_size;
  if (ws_size < O_END) return;

  const float* map_rep = (const float*)d_in[0];
  const float* gfeat   = (const float*)d_in[1];
  const float* memory  = (const float*)d_in[2];
  const int*   fog     = (const int*)d_in[3];
  const float* hex_w = (const float*)d_in[4];  const float* hex_b = (const float*)d_in[5];
  const float* glob_w = (const float*)d_in[6]; const float* glob_b = (const float*)d_in[7];
  const float* mem_wq = (const float*)d_in[8];  const float* mem_bq = (const float*)d_in[9];
  const float* mem_wk = (const float*)d_in[10]; const float* mem_bk = (const float*)d_in[11];
  const float* mem_wv = (const float*)d_in[12]; const float* mem_bv = (const float*)d_in[13];
  const float* mem_wo = (const float*)d_in[14]; const float* mem_bo = (const float*)d_in[15];
  const float* gate_w = (const float*)d_in[16]; const float* gate_b = (const float*)d_in[17];
  const float* mn1_g = (const float*)d_in[18];  const float* mn1_b = (const float*)d_in[19];
  const float* mn2_g = (const float*)d_in[20];  const float* mn2_b = (const float*)d_in[21];
  const float* lw_q = (const float*)d_in[22];   const float* lb_q = (const float*)d_in[23];
  const float* lw_k = (const float*)d_in[24];   const float* lb_k = (const float*)d_in[25];
  const float* lw_v = (const float*)d_in[26];   const float* lb_v = (const float*)d_in[27];
  const float* lw_o = (const float*)d_in[28];   const float* lb_o = (const float*)d_in[29];
  const float* ln1_g = (const float*)d_in[30];  const float* ln1_b = (const float*)d_in[31];
  const float* ln2_g = (const float*)d_in[32];  const float* ln2_b = (const float*)d_in[33];
  const float* ff1_w = (const float*)d_in[34];  const float* ff1_b = (const float*)d_in[35];
  const float* ff2_w = (const float*)d_in[36];  const float* ff2_b = (const float*)d_in[37];
  const float* pool_w = (const float*)d_in[38]; const float* pool_b = (const float*)d_in[39];

  char* ws = (char*)d_ws;
  float*  xf     = (float*)(ws + O_XF);
  float*  tmpf   = (float*)(ws + O_TMPF);
  float*  seqf   = (float*)(ws + O_SEQF);
  bf16_t* xb     = (bf16_t*)(ws + O_XB);
  bf16_t* seqb   = (bf16_t*)(ws + O_SEQB);
  bf16_t* qb     = (bf16_t*)(ws + O_QB);
  bf16_t* kb     = (bf16_t*)(ws + O_KB);
  bf16_t* vb     = (bf16_t*)(ws + O_VB);
  bf16_t* vtb    = (bf16_t*)(ws + O_VTB);
  bf16_t* attnb  = (bf16_t*)(ws + O_ATTNB);
  bf16_t* hb     = (bf16_t*)(ws + O_HB);
  bf16_t* mapb   = (bf16_t*)(ws + O_MAPB);
  bf16_t* memb   = (bf16_t*)(ws + O_MEMB);
  bf16_t* qmemb  = (bf16_t*)(ws + O_QMEMB);
  bf16_t* kmemb  = (bf16_t*)(ws + O_KMEMB);
  bf16_t* vmemb  = (bf16_t*)(ws + O_VMEMB);
  bf16_t* vtmemb = (bf16_t*)(ws + O_VTMEMB);
  bf16_t* mupatt = (bf16_t*)(ws + O_MUPATT);
  float*  mupdf  = (float*)(ws + O_MUPDF);
  float*  gatef  = (float*)(ws + O_GATEF);
  bf16_t* catb   = (bf16_t*)(ws + O_CATB);
  float*  globf  = (float*)(ws + O_GLOBF);
  float*  logits = (float*)(ws + O_LOGITS);
  float*  pooled = (float*)(ws + O_POOLED);
  bf16_t* whext  = (bf16_t*)(ws + O_WHEXT);
  bf16_t* wmqt   = (bf16_t*)(ws + O_WMQT);
  bf16_t* wmkt   = (bf16_t*)(ws + O_WMKT);
  bf16_t* wmvt   = (bf16_t*)(ws + O_WMVT);
  bf16_t* wmot   = (bf16_t*)(ws + O_WMOT);
  bf16_t* wgatet = (bf16_t*)(ws + O_WGATET);
  bf16_t* wlqt   = (bf16_t*)(ws + O_WLQT);
  bf16_t* wlkt   = (bf16_t*)(ws + O_WLKT);
  bf16_t* wlvt   = (bf16_t*)(ws + O_WLVT);
  bf16_t* wlot   = (bf16_t*)(ws + O_WLOT);
  bf16_t* wf1t   = (bf16_t*)(ws + O_WF1T);
  bf16_t* wf2t   = (bf16_t*)(ws + O_WF2T);
  float* out = (float*)d_out;

  const int MS = Bn * Sn;   // 4608
  const int MM = Bn * Mn;   // 256
  const float scale = 0.125f;   // 1/sqrt(64)

  // ---- weight conversion (transposed bf16) ----
  wconv(stream, hex_w, whext, FHEX, Dn, FHEXP);
  wconv(stream, mem_wq, wmqt, Dn, Dn, Dn);
  wconv(stream, mem_wk, wmkt, Dn, Dn, Dn);
  wconv(stream, mem_wv, wmvt, Dn, Dn, Dn);
  wconv(stream, mem_wo, wmot, Dn, Dn, Dn);
  wconv(stream, gate_w, wgatet, 2 * Dn, Dn, 2 * Dn);

  // ---- embed: seq = map@hex_w + hex_b + glob ----
  k_glob<<<(Bn * Dn + 255) / 256, 256, 0, stream>>>(gfeat, glob_w, glob_b, globf);
  {
    long total = (long)MS * FHEXP;
    k_aconv<<<(total + 255) / 256, 256, 0, stream>>>(map_rep, mapb, MS, FHEX, FHEXP);
  }
  gemm(stream, mapb, FHEXP, whext, FHEXP, hex_b, nullptr, 0, globf, Sn,
       seqf, seqb, Dn, MS, Dn, FHEXP, GF_BIAS | GF_ROWVEC);

  // ---- memory projections ----
  {
    long total = (long)MM * Dn;
    k_aconv<<<(total + 255) / 256, 256, 0, stream>>>(memory, memb, MM, Dn, Dn);
  }
  gemm(stream, seqb, Dn, wmqt, Dn, mem_bq, nullptr, 0, nullptr, 0,
       nullptr, qb, Dn, MS, Dn, Dn, GF_BIAS);                       // Q(seq)
  gemm(stream, memb, Dn, wmkt, Dn, mem_bk, nullptr, 0, nullptr, 0,
       nullptr, kmemb, Dn, MM, Dn, Dn, GF_BIAS);                    // K(mem)
  gemm(stream, memb, Dn, wmvt, Dn, mem_bv, nullptr, 0, nullptr, 0,
       nullptr, vmemb, Dn, MM, Dn, Dn, GF_BIAS);                    // V(mem)
  {
    long total = (long)Bn * Mn * Dn;
    k_btrans<<<(total + 255) / 256, 256, 0, stream>>>(vmemb, vtmemb, Bn, Mn, Dn);
  }
  // ---- cross attention seq -> memory ----
  k_flash<<<dim3(Sn / 64, NHn, Bn), 128, 0, stream>>>(qb, kmemb, vtmemb, nullptr,
                                                      attnb, Sn, Mn, scale);
  gemm(stream, attnb, Dn, wmot, Dn, mem_bo, seqf, Dn, nullptr, 0,
       tmpf, nullptr, Dn, MS, Dn, Dn, GF_BIAS | GF_RES);
  k_ln<<<MS, 256, 0, stream>>>(tmpf, mn1_g, mn1_b, xf, xb);

  // ---- memory update: mem_upd = MHA(memory, seq, seq) ----
  gemm(stream, seqb, Dn, wmkt, Dn, mem_bk, nullptr, 0, nullptr, 0,
       nullptr, kb, Dn, MS, Dn, Dn, GF_BIAS);                       // K(seq)
  gemm(stream, seqb, Dn, wmvt, Dn, mem_bv, nullptr, 0, nullptr, 0,
       nullptr, vb, Dn, MS, Dn, Dn, GF_BIAS);                       // V(seq)
  {
    long total = (long)Bn * Sn * Dn;
    k_btrans<<<(total + 255) / 256, 256, 0, stream>>>(vb, vtb, Bn, Sn, Dn);
  }
  gemm(stream, memb, Dn, wmqt, Dn, mem_bq, nullptr, 0, nullptr, 0,
       nullptr, qmemb, Dn, MM, Dn, Dn, GF_BIAS);                    // Q(mem)
  k_flash<<<dim3(Mn / 64, NHn, Bn), 128, 0, stream>>>(qmemb, kb, vtb, nullptr,
                                                      mupatt, Mn, Sn, scale);
  gemm(stream, mupatt, Dn, wmot, Dn, mem_bo, nullptr, 0, nullptr, 0,
       mupdf, nullptr, Dn, MM, Dn, Dn, GF_BIAS);
  {
    long total = (long)MM * 1024;
    k_cat<<<(total + 255) / 256, 256, 0, stream>>>(memory, mupdf, catb);
  }
  gemm(stream, catb, 2 * Dn, wgatet, 2 * Dn, gate_b, nullptr, 0, nullptr, 0,
       gatef, nullptr, Dn, MM, Dn, 2 * Dn, GF_BIAS | GF_SIGM);
  k_newmem<<<MM, 256, 0, stream>>>(gatef, mupdf, memory, mn2_g, mn2_b, out + 444);

  // ---- 4 transformer layers ----
  for (int l = 0; l < Ln; ++l) {
    const long wsz = (long)Dn * Dn;
    wconv(stream, lw_q + l * wsz, wlqt, Dn, Dn, Dn);
    wconv(stream, lw_k + l * wsz, wlkt, Dn, Dn, Dn);
    wconv(stream, lw_v + l * wsz, wlvt, Dn, Dn, Dn);
    wconv(stream, lw_o + l * wsz, wlot, Dn, Dn, Dn);
    wconv(stream, ff1_w + (long)l * Dn * DFn, wf1t, Dn, DFn, Dn);
    wconv(stream, ff2_w + (long)l * DFn * Dn, wf2t, DFn, Dn, DFn);

    gemm(stream, xb, Dn, wlqt, Dn, lb_q + l * Dn, nullptr, 0, nullptr, 0,
         nullptr, qb, Dn, MS, Dn, Dn, GF_BIAS);
    gemm(stream, xb, Dn, wlkt, Dn, lb_k + l * Dn, nullptr, 0, nullptr, 0,
         nullptr, kb, Dn, MS, Dn, Dn, GF_BIAS);
    gemm(stream, xb, Dn, wlvt, Dn, lb_v + l * Dn, nullptr, 0, nullptr, 0,
         nullptr, vb, Dn, MS, Dn, Dn, GF_BIAS);
    {
      long total = (long)Bn * Sn * Dn;
      k_btrans<<<(total + 255) / 256, 256, 0, stream>>>(vb, vtb, Bn, Sn, Dn);
    }
    k_flash<<<dim3(Sn / 64, NHn, Bn), 128, 0, stream>>>(qb, kb, vtb, fog,
                                                        attnb, Sn, Sn, scale);
    gemm(stream, attnb, Dn, wlot, Dn, lb_o + l * Dn, xf, Dn, nullptr, 0,
         tmpf, nullptr, Dn, MS, Dn, Dn, GF_BIAS | GF_RES);
    k_ln<<<MS, 256, 0, stream>>>(tmpf, ln1_g + l * Dn, ln1_b + l * Dn, xf, xb);

    gemm(stream, xb, Dn, wf1t, Dn, ff1_b + l * DFn, nullptr, 0, nullptr, 0,
         nullptr, hb, DFn, MS, DFn, Dn, GF_BIAS | GF_GELU);
    gemm(stream, hb, DFn, wf2t, DFn, ff2_b + l * Dn, xf, Dn, nullptr, 0,
         tmpf, nullptr, Dn, MS, Dn, DFn, GF_BIAS | GF_RES);
    k_ln<<<MS, 256, 0, stream>>>(tmpf, ln2_g + l * Dn, ln2_b + l * Dn, xf, xb);
  }

  // ---- attention pooling + heads ----
  k_logits<<<(MS + 255) / 256, 256, 0, stream>>>(xf, pool_w, pool_b, logits, MS);
  k_pool<<<Bn, 256, 0, stream>>>(xf, logits, pooled);
  k_heads<<<1, 512, 0, stream>>>(pooled,
      (const float*)d_in[40], (const float*)d_in[41], (const float*)d_in[42], (const float*)d_in[43],
      (const float*)d_in[44], (const float*)d_in[45], (const float*)d_in[46], (const float*)d_in[47],
      (const float*)d_in[48], (const float*)d_in[49], (const float*)d_in[50], (const float*)d_in[51],
      (const float*)d_in[52], (const float*)d_in[53], (const float*)d_in[54], (const float*)d_in[55],
      out);
}
